// AudioModel_67705864454828
// MI455X (gfx1250) — matrix operand order repaired and checked by reading, not compiled, not run on hardware
//
#include <hip/hip_runtime.h>
#include <hip/hip_bf16.h>

typedef __attribute__((ext_vector_type(16))) _Float16 v16h;
typedef __attribute__((ext_vector_type(8)))  _Float16 v8h;
typedef __attribute__((ext_vector_type(8)))  float    v8f;

#define PI_F 3.14159265358979323846f
#define LOWEST_FREQ (30.0f/11025.0f)

// ---------------- workspace layout (bytes, all 256B aligned) ----------------
enum : size_t {
  OFF_XF16  = 0,                      // 16*128 h
  OFF_WLIN  = OFF_XF16  + 4096,       // 512*128 h
  OFF_WUP   = OFF_WLIN  + 131072,     // 3*128*384 h
  OFF_WAMP  = OFF_WUP   + 294912,     // 128*128 h
  OFF_WFREQ = OFF_WAMP  + 32768,      // 128*128 h
  OFF_WNZ   = OFF_WFREQ + 32768,      // 4*128*384 h
  OFF_WOUT0 = OFF_WNZ   + 393216,     // 384 h
  OFF_ACT0  = OFF_WOUT0 + 1024,       // 16*4*128 h
  OFF_U1    = OFF_ACT0  + 16384,      // T=8
  OFF_U2    = OFF_U1    + 32768,      // T=16
  OFF_U3    = OFF_U2    + 65536,      // T=32  (= h, control signal)
  OFF_N1    = OFF_U3    + 131072,     // T=64
  OFF_N2    = OFF_N1    + 262144,     // T=128
  OFF_N3    = OFF_N2    + 524288,     // T=256
  OFF_N4    = OFF_N3    + 1048576,    // T=512
  OFF_AMPC  = OFF_N4    + 2097152,    // 16*32*128 f32 [b][t][o]
  OFF_FREQC = OFF_AMPC  + 262144,
  OFF_BASE  = OFF_FREQC + 262144,     // 16*32*128 f32 [b][m][o]
  OFF_CDC   = OFF_BASE  + 262144      // 16*1024 f32 frame DC constants
};

// ---------------- WMMA fragment helpers ----------------
// A (16x32 f16), operand stored row-major [M][K]:
//  lanes 0-15 : M=lane,   halfs 0-7 -> K=kk0+0..7,  halfs 8-15 -> K=kk0+16..23
//  lanes16-31 : M=lane-16,halfs 0-7 -> K=kk0+8..15, halfs 8-15 -> K=kk0+24..31
__device__ inline v16h load_fragA(const _Float16* W, int ldk, int m0, int kk0, int lane) {
  int r  = lane & 15;
  int hs = ((lane >> 4) & 1) << 3;            // 0 or 8 halfs
  const _Float16* p = W + (size_t)(m0 + r) * ldk + kk0 + hs;
  v8h lo = *(const v8h*)p;
  v8h hi = *(const v8h*)(p + 16);
  v16h f;
#pragma unroll
  for (int i = 0; i < 8; ++i) { f[i] = lo[i]; f[i + 8] = hi[i]; }
  return f;
}

// B (32x16 f16), operand stored row-major [N][K] (i.e. B^T):
//  lanes 0-15: N=lane, K=kk0+0..7,16..23 ; lanes 16-31: N=lane-16, K=kk0+8..15,24..31
__device__ inline v16h load_fragB(const _Float16* base, int rowStride, int kk0, int lane) {
  int r  = lane & 15;
  int ks = ((lane >> 4) & 1) << 3;            // 0 or 8 halfs
  const _Float16* p = base + (size_t)r * rowStride + kk0 + ks;
  v8h lo = *(const v8h*)p;
  v8h hi = *(const v8h*)(p + 16);
  v16h f;
#pragma unroll
  for (int i = 0; i < 8; ++i) { f[i] = lo[i]; f[i + 8] = hi[i]; }
  return f;
}

__device__ inline v8f wmma_f16(v16h a, v16h b, v8f c) {
  return __builtin_amdgcn_wmma_f32_16x16x32_f16(false, a, false, b, (short)0, c, false, false);
}

// ---------------- kernel 1: convert inputs/weights to f16 layouts ----------------
__global__ void k_prep(const float* x, const float* wlin, const float* wup,
                       const float* wamp, const float* wfreq,
                       const float* wnz, const float* wout,
                       _Float16* dx, _Float16* dlin, _Float16* dup,
                       _Float16* damp, _Float16* dfreq, _Float16* dnz, _Float16* dout0) {
  int i = blockIdx.x * blockDim.x + threadIdx.x;
  int stride = gridDim.x * blockDim.x;
  for (int idx = i; idx < 16 * 128; idx += stride) dx[idx] = (_Float16)x[idx];
  for (int idx = i; idx < 512 * 128; idx += stride) dlin[idx] = (_Float16)wlin[idx];
  for (int idx = i; idx < 3 * 128 * 384; idx += stride) {   // [l][o][j*128+c] <- [l][o][c][j]
    int l = idx / 49152, r = idx % 49152;
    int o = r / 384, kk = r % 384, j = kk >> 7, c = kk & 127;
    dup[idx] = (_Float16)wup[(((size_t)l * 128 + o) * 128 + c) * 3 + j];
  }
  for (int idx = i; idx < 128 * 128; idx += stride) damp[idx] = (_Float16)wamp[idx];
  for (int idx = i; idx < 128 * 128; idx += stride) dfreq[idx] = (_Float16)wfreq[idx];
  for (int idx = i; idx < 4 * 128 * 384; idx += stride) {
    int l = idx / 49152, r = idx % 49152;
    int o = r / 384, kk = r % 384, j = kk >> 7, c = kk & 127;
    dnz[idx] = (_Float16)wnz[(((size_t)l * 128 + o) * 128 + c) * 3 + j];
  }
  for (int idx = i; idx < 384; idx += stride) {              // only output channel 0 needed
    int j = idx >> 7, c = idx & 127;
    dout0[idx] = (_Float16)wout[c * 3 + j];
  }
}

// ---------------- kernel 2: up_lin GEMM  (M=512, N=16 batches, K=128) ----------------
__global__ void k_gemm_lin(const _Float16* __restrict__ W, const _Float16* __restrict__ X,
                           const float* __restrict__ bias, _Float16* __restrict__ act0) {
  int lane = threadIdx.x & 31;
  int wv = threadIdx.x >> 5;
  int m0 = (blockIdx.x * 4 + wv) * 16;
  v8f acc = {};
#pragma unroll
  for (int ks = 0; ks < 4; ++ks) {
    v16h a = load_fragA(W, 128, m0, ks * 32, lane);
    v16h b = load_fragB(X, 128, ks * 32, lane);
    acc = wmma_f16(a, b, acc);
  }
  int bb = lane & 15;
  int mbase = m0 + (((lane >> 4) & 1) << 3);
#pragma unroll
  for (int i = 0; i < 8; ++i) {
    int m = mbase + i;
    float v = acc[i] + bias[m];
    int c = m >> 2, t = m & 3;                    // h0 reshape (B,128,4)
    act0[((size_t)bb * 4 + t) * 128 + c] = (_Float16)v;  // store [b][t][c]
  }
}

// ------- kernel 3: repeat-x2 + conv1d(k=3,'same') + leaky, as GEMM  -------
// act layout [b][t][c]; weight [o][j*128+c]; grid = (2, Tout), block 128 (4 waves)
__global__ void k_conv(const _Float16* __restrict__ actIn, const _Float16* __restrict__ W,
                       const float* __restrict__ bias, _Float16* __restrict__ actOut, int Tout) {
  int lane = threadIdx.x & 31;
  int wv = threadIdx.x >> 5;
  int m0 = (blockIdx.x * 4 + wv) * 16;
  int t = blockIdx.y;
  int Tin = Tout >> 1;
  v8f acc = {};
#pragma unroll
  for (int ks = 0; ks < 12; ++ks) {
    int kk0 = ks * 32;
    int j = kk0 >> 7, c0 = kk0 & 127;
    int tt = t + j - 1;
    v16h a = load_fragA(W, 384, m0, kk0, lane);
    v16h b;
    if (tt < 0 || tt >= Tout) {
#pragma unroll
      for (int i = 0; i < 16; ++i) b[i] = (_Float16)0.0f;
    } else {
      int tin = tt >> 1;                                   // nearest-x2 upsample
      b = load_fragB(actIn + (size_t)tin * 128, Tin * 128, c0, lane);
    }
    acc = wmma_f16(a, b, acc);
  }
  int bb = lane & 15;
  int obase = m0 + (((lane >> 4) & 1) << 3);
  v8h o8;
#pragma unroll
  for (int i = 0; i < 8; ++i) {
    float v = acc[i] + bias[obase + i];
    v = v >= 0.0f ? v : 0.2f * v;                          // leaky(0.2)
    o8[i] = (_Float16)v;
  }
  *(v8h*)(actOut + ((size_t)bb * Tout + t) * 128 + obase) = o8;
}

// ------- kernel 4: oscillator projections (amp & freq share B fragments) -------
__global__ void k_osc_proj(const _Float16* __restrict__ H, const _Float16* __restrict__ Wa,
                           const _Float16* __restrict__ Wf, const float* __restrict__ ba,
                           const float* __restrict__ bf, float* __restrict__ ampc,
                           float* __restrict__ freqc) {
  int lane = threadIdx.x & 31;
  int wv = threadIdx.x >> 5;
  int m0 = (blockIdx.x * 4 + wv) * 16;
  int t = blockIdx.y;                                      // 0..31
  v8f accA = {}, accF = {};
#pragma unroll
  for (int ks = 0; ks < 4; ++ks) {
    int kk0 = ks * 32;
    v16h b  = load_fragB(H + (size_t)t * 128, 32 * 128, kk0, lane);
    v16h aa = load_fragA(Wa, 128, m0, kk0, lane);
    v16h af = load_fragA(Wf, 128, m0, kk0, lane);
    accA = wmma_f16(aa, b, accA);
    accF = wmma_f16(af, b, accF);
  }
  int bb = lane & 15;
  int obase = m0 + (((lane >> 4) & 1) << 3);
#pragma unroll
  for (int i = 0; i < 8; ++i) {
    int o = obase + i;
    float va = accA[i] + ba[o];
    va = va * va;                                          // amp_activation = x^2
    float s = 1.0f / (1.0f + __expf(-(accF[i] + bf[o])));
    float vf = LOWEST_FREQ + s * (1.0f - LOWEST_FREQ);
    ampc [((size_t)bb * 32 + t) * 128 + o] = va;
    freqc[((size_t)bb * 32 + t) * 128 + o] = vf;
  }
}

// ------- kernel 5: per-(b,o) phase base at each 1024-sample block (closed-form cumsum) -------
__global__ void k_osc_base(const float* __restrict__ freqc, float* __restrict__ base) {
  int idx = blockIdx.x * blockDim.x + threadIdx.x;
  if (idx >= 16 * 128) return;
  int b = idx >> 7, o = idx & 127;
  float xv[32];
#pragma unroll
  for (int t = 0; t < 32; ++t) xv[t] = freqc[((size_t)b * 32 + t) * 128 + o];
  float S = 0.0f;
#pragma unroll
  for (int m = 0; m < 32; ++m) {
    base[((size_t)b * 32 + m) * 128 + o] = PI_F * S;
    float fh = (m == 0)  ? 512.0f * xv[0]  : 128.0f * xv[m - 1] + 384.0f * xv[m];
    float sh = (m == 31) ? 512.0f * xv[31] : 384.0f * xv[m] + 128.0f * xv[m + 1];
    S += fh + sh;
  }
}

// ------- kernel 6: noise spectral DC per frame (mask_after=1 => only k=0 survives) -------
__global__ void k_spec0(const _Float16* __restrict__ s4, const _Float16* __restrict__ W0,
                        const float* __restrict__ b0, const float* __restrict__ wn,
                        float* __restrict__ cdc) {
  int idx = blockIdx.x * blockDim.x + threadIdx.x;
  if (idx >= 16 * 1024) return;
  int b = idx >> 10, t = idx & 1023;
  float acc = 0.0f;
#pragma unroll
  for (int j = 0; j < 3; ++j) {
    int tt = t + j - 1;
    if (tt < 0 || tt >= 1024) continue;
    int tin = tt >> 1;                                     // final repeat-x2
    const _Float16* p = s4 + ((size_t)b * 512 + tin) * 128;
    const _Float16* w = W0 + j * 128;
#pragma unroll 8
    for (int c = 0; c < 128; ++c) acc += (float)w[c] * (float)p[c];
  }
  acc += b0[0];
  float spec = acc * acc;                                  // squared=True
  const float* f = wn + ((size_t)b * 1024 + t) * 64;
  float wsum = 0.0f;
#pragma unroll
  for (int n = 0; n < 64; ++n) {
    float win = 0.5f * (1.0f - __cosf((2.0f * PI_F / 64.0f) * (float)n));
    wsum += (2.0f * f[n] - 1.0f) * win;                    // X[0] of rfft(wn*win)
  }
  cdc[(size_t)b * 1024 + t] = spec * wsum * (1.0f / 64.0f); // irfft DC => constant frame
}

// ------- kernel 7: oscillator bank synthesis + fused noise overlap-add -------
// grid (m=32 blocks of 1024 samples, b=16), block 256 threads x 4 samples
__global__ void k_synth(const float* __restrict__ ampc, const float* __restrict__ freqc,
                        const float* __restrict__ basec, const float* __restrict__ cdc,
                        float* __restrict__ out) {
  int m = blockIdx.x, b = blockIdx.y;
  __shared__ float sF[3 * 128], sA[3 * 128], sB[128];
  int tid = threadIdx.x;
  if (tid < 128) {
    int o = tid;
    int mm1 = m > 0 ? m - 1 : 0, mp1 = m < 31 ? m + 1 : 31;
    sF[o]       = freqc[((size_t)b * 32 + mm1) * 128 + o];
    sF[128 + o] = freqc[((size_t)b * 32 + m)   * 128 + o];
    sF[256 + o] = freqc[((size_t)b * 32 + mp1) * 128 + o];
    sA[o]       = ampc [((size_t)b * 32 + mm1) * 128 + o];
    sA[128 + o] = ampc [((size_t)b * 32 + m)   * 128 + o];
    sA[256 + o] = ampc [((size_t)b * 32 + mp1) * 128 + o];
    sB[o]       = basec[((size_t)b * 32 + m)   * 128 + o];
  }
  __syncthreads();
  float h[4] = {0.0f, 0.0f, 0.0f, 0.0f};
  int L0 = tid * 4;
  for (int o = 0; o < 128; ++o) {
    float fa = sF[o], fb = sF[128 + o], fc = sF[256 + o];
    float aa = sA[o], ab = sA[128 + o], ac = sA[256 + o];
    float bs = sB[o];
    float FH = (m == 0) ? 512.0f * fb : 512.0f * fa + 384.0f * (fb - fa);
#pragma unroll
    for (int u = 0; u < 4; ++u) {
      int L = L0 + u;
      float P, av;
      if (L < 512) {                                        // segment (m-1 -> m)
        float n = (float)(L + 1);
        if (m == 0) { P = n * fb; av = ab; }                // pos clipped to 0
        else {
          P = n * fa + (fb - fa) * (0.5f * n + n * n * (1.0f / 2048.0f));
          float w = 0.5f + ((float)L + 0.5f) * (1.0f / 1024.0f);
          av = aa + (ab - aa) * w;
        }
      } else {                                              // segment (m -> m+1)
        float n2 = (float)(L - 511);
        if (m == 31) { P = FH + n2 * fb; av = ab; }         // pos clipped to 31
        else {
          P = FH + n2 * fb + (fc - fb) * (n2 * n2 * (1.0f / 2048.0f));
          float w2 = ((float)(L - 512) + 0.5f) * (1.0f / 1024.0f);
          av = ab + (ac - ab) * w2;
        }
      }
      float ph = fmaf(PI_F, P, bs);
      h[u] += __sinf(ph) * av;
    }
  }
#pragma unroll
  for (int u = 0; u < 4; ++u) {
    int s = m * 1024 + L0 + u;
    int t0 = s >> 5;                                        // overlap-add of constant frames
    float nz = cdc[(size_t)b * 1024 + t0] + (t0 > 0 ? cdc[(size_t)b * 1024 + t0 - 1] : 0.0f);
    out[(size_t)b * 32768 + s] = h[u] + nz;
  }
}

// ---------------- host launch ----------------
extern "C" void kernel_launch(void* const* d_in, const int* in_sizes, int n_in,
                              void* d_out, int out_size, void* d_ws, size_t ws_size,
                              hipStream_t stream) {
  const float* x     = (const float*)d_in[0];
  const float* wn    = (const float*)d_in[1];
  const float* wlin  = (const float*)d_in[2];
  const float* blin  = (const float*)d_in[3];
  const float* wup   = (const float*)d_in[4];
  const float* bup   = (const float*)d_in[5];
  const float* wamp  = (const float*)d_in[6];
  const float* bamp  = (const float*)d_in[7];
  const float* wfreq = (const float*)d_in[8];
  const float* bfreq = (const float*)d_in[9];
  const float* wnz   = (const float*)d_in[10];
  const float* bnz   = (const float*)d_in[11];
  const float* wout  = (const float*)d_in[12];
  const float* bout  = (const float*)d_in[13];
  float* out = (float*)d_out;
  char* ws = (char*)d_ws;

  _Float16* dx    = (_Float16*)(ws + OFF_XF16);
  _Float16* dlin  = (_Float16*)(ws + OFF_WLIN);
  _Float16* dup   = (_Float16*)(ws + OFF_WUP);
  _Float16* damp  = (_Float16*)(ws + OFF_WAMP);
  _Float16* dfreq = (_Float16*)(ws + OFF_WFREQ);
  _Float16* dnz   = (_Float16*)(ws + OFF_WNZ);
  _Float16* dout0 = (_Float16*)(ws + OFF_WOUT0);
  _Float16* act0  = (_Float16*)(ws + OFF_ACT0);
  _Float16* u1    = (_Float16*)(ws + OFF_U1);
  _Float16* u2    = (_Float16*)(ws + OFF_U2);
  _Float16* u3    = (_Float16*)(ws + OFF_U3);
  _Float16* n1    = (_Float16*)(ws + OFF_N1);
  _Float16* n2    = (_Float16*)(ws + OFF_N2);
  _Float16* n3    = (_Float16*)(ws + OFF_N3);
  _Float16* n4    = (_Float16*)(ws + OFF_N4);
  float* ampc  = (float*)(ws + OFF_AMPC);
  float* freqc = (float*)(ws + OFF_FREQC);
  float* basec = (float*)(ws + OFF_BASE);
  float* cdc   = (float*)(ws + OFF_CDC);

  k_prep<<<128, 256, 0, stream>>>(x, wlin, wup, wamp, wfreq, wnz, wout,
                                  dx, dlin, dup, damp, dfreq, dnz, dout0);
  k_gemm_lin<<<8, 128, 0, stream>>>(dlin, dx, blin, act0);

  k_conv<<<dim3(2, 8),   128, 0, stream>>>(act0, dup + 0 * 49152, bup + 0,   u1, 8);
  k_conv<<<dim3(2, 16),  128, 0, stream>>>(u1,   dup + 1 * 49152, bup + 128, u2, 16);
  k_conv<<<dim3(2, 32),  128, 0, stream>>>(u2,   dup + 2 * 49152, bup + 256, u3, 32);

  k_conv<<<dim3(2, 64),  128, 0, stream>>>(u3, dnz + 0 * 49152, bnz + 0,   n1, 64);
  k_conv<<<dim3(2, 128), 128, 0, stream>>>(n1, dnz + 1 * 49152, bnz + 128, n2, 128);
  k_conv<<<dim3(2, 256), 128, 0, stream>>>(n2, dnz + 2 * 49152, bnz + 256, n3, 256);
  k_conv<<<dim3(2, 512), 128, 0, stream>>>(n3, dnz + 3 * 49152, bnz + 384, n4, 512);

  k_osc_proj<<<dim3(2, 32), 128, 0, stream>>>(u3, damp, dfreq, bamp, bfreq, ampc, freqc);
  k_osc_base<<<8, 256, 0, stream>>>(freqc, basec);
  k_spec0<<<64, 256, 0, stream>>>(n4, dout0, bout, wn, cdc);
  k_synth<<<dim3(32, 16), 256, 0, stream>>>(ampc, freqc, basec, cdc, out);
}
